// TaxoGNN_7971459301988
// MI455X (gfx1250) — matrix operand order repaired and checked by reading, not compiled, not run on hardware
//
#include <hip/hip_runtime.h>

#define ETA   0.5f
#define SLOPE 0.01f
#define ENC_NEG_INF 0x007FFFFFu   // ordered-uint encoding of -inf

typedef __attribute__((ext_vector_type(16))) _Float16 v16h;
typedef __attribute__((ext_vector_type(8)))  float    v8f;

typedef int v4i __attribute__((vector_size(16)));
typedef __attribute__((address_space(1))) void as1_void;
typedef __attribute__((address_space(3))) void as3_void;
typedef __attribute__((address_space(1))) v4i  as1_v4i;
typedef __attribute__((address_space(3))) v4i  as3_v4i;

#if defined(__has_builtin)
# if __has_builtin(__builtin_amdgcn_global_load_async_to_lds_b128)
#  define HAVE_ASYNC_LDS 1
# endif
#endif
#ifndef HAVE_ASYNC_LDS
# define HAVE_ASYNC_LDS 0
#endif

// ---- monotone float <-> ordered uint (for atomic max on floats) ----
__device__ __forceinline__ unsigned f2ord(float f) {
  unsigned u = __float_as_uint(f);
  return (u & 0x80000000u) ? ~u : (u | 0x80000000u);
}
__device__ __forceinline__ float ord2f(unsigned u) {
  return __uint_as_float((u & 0x80000000u) ? (u & 0x7fffffffu) : ~u);
}
__device__ __forceinline__ float wred(float v) {
  #pragma unroll
  for (int o = 16; o > 0; o >>= 1) v += __shfl_xor(v, o, 32);
  return v;
}

// =====================================================================
// WMMA GEMM: out(N x NCOLS) = A(N x 64) @ B + bias, optional PReLU.
// TRANSB=true : B = W^T, W row-major (NCOLS x 64)  -> B[k][n] = W[n*64+k]
// TRANSB=false: B = W,   W row-major (64 x NCOLS)  -> B[k][n] = W[k*NCOLS+n]
// 256 threads = 8 waves; each wave owns a 16-row M-tile; block covers 128 rows.
// A tile staged to LDS via async global->LDS (ASYNCcnt) when available;
// W staged as f16 (converted once). Math: v_wmma_f32_16x16x32_f16.
// =====================================================================
template<int NCOLS, bool TRANSB, bool PRELU>
__global__ void __launch_bounds__(256)
k_gemm64(const float* __restrict__ A, const float* __restrict__ W,
         const float* __restrict__ bias, const float* __restrict__ pa,
         float* __restrict__ out, int nrows)
{
  __shared__ float    Asf[128 * 64];
  __shared__ _Float16 Wsh[NCOLS * 64];
  const int tid  = threadIdx.x;
  const int lane = tid & 31;
  const int wave = tid >> 5;
  const int rowblock = blockIdx.x * 128;

#if HAVE_ASYNC_LDS
  // 2048 float4 chunks; each lane DMAs 16B global -> LDS (no VGPR round trip)
  #pragma unroll
  for (int it = 0; it < 8; ++it) {
    int q = tid + it * 256;                 // float4 index in 128x64 tile
    int r = rowblock + (q >> 4);
    if (r < nrows) {
      __builtin_amdgcn_global_load_async_to_lds_b128(
          (as1_v4i*)(as1_void*)(void*)(A + (size_t)r * 64 + (size_t)(q & 15) * 4),
          (as3_v4i*)(as3_void*)(void*)(Asf + (size_t)q * 4), 0, 0);
    }
  }
#else
  for (int i = tid; i < 128 * 64; i += 256) {
    int r = rowblock + (i >> 6);
    Asf[i] = (r < nrows) ? A[(size_t)r * 64 + (i & 63)] : 0.0f;
  }
#endif
  for (int i = tid; i < NCOLS * 64; i += 256) Wsh[i] = (_Float16)W[i];
#if HAVE_ASYNC_LDS
# if defined(__has_builtin) && __has_builtin(__builtin_amdgcn_s_wait_asynccnt)
  __builtin_amdgcn_s_wait_asynccnt(0);
# else
  asm volatile("s_wait_asynccnt 0x0" ::: "memory");
# endif
#endif
  __syncthreads();

  const int half = lane >> 4;
  const int l16  = lane & 15;

  // A fragments (16x32 f16 layout): lanes0-15 K={0..7,16..23}; lanes16-31 K={8..15,24..31}
  v16h afrag[2];
  #pragma unroll
  for (int kc = 0; kc < 2; ++kc) {
    #pragma unroll
    for (int j = 0; j < 16; ++j) {
      int K = kc * 32 + ((j < 8) ? (half * 8 + j) : (16 + half * 8 + (j - 8)));
      afrag[kc][j] = (_Float16)Asf[(wave * 16 + l16) * 64 + K];
    }
  }

  constexpr int NT = NCOLS / 16;
  v8f acc[NT];
  #pragma unroll
  for (int nt = 0; nt < NT; ++nt) {
    v8f c = {};
    #pragma unroll
    for (int kc = 0; kc < 2; ++kc) {
      v16h bfrag;
      #pragma unroll
      for (int j = 0; j < 16; ++j) {
        int K = kc * 32 + ((j < 8) ? (half * 8 + j) : (16 + half * 8 + (j - 8)));    // B (32x16): lanes0-15 K={0..7,16..23}, lanes16-31 K={8..15,24..31}
        int n = nt * 16 + l16;
        bfrag[j] = TRANSB ? Wsh[n * 64 + K] : Wsh[K * NCOLS + n];
      }
      c = __builtin_amdgcn_wmma_f32_16x16x32_f16(false, afrag[kc], false, bfrag,
                                                 (short)0, c, false, false);
    }
    acc[nt] = c;
  }

  float a0 = 0.0f;
  if (PRELU) a0 = pa[0];
  #pragma unroll
  for (int nt = 0; nt < NT; ++nt) {
    int n = nt * 16 + l16;
    float bv = bias ? bias[n] : 0.0f;
    #pragma unroll
    for (int r = 0; r < 8; ++r) {
      int row = rowblock + wave * 16 + half * 8 + r;   // C: M = half*8 + reg
      if (row < nrows) {
        float v = acc[nt][r] + bv;
        if (PRELU) v = (v >= 0.0f) ? v : a0 * v;
        out[(size_t)row * NCOLS + n] = v;
      }
    }
  }
}

// =====================================================================
// Elementwise / gather / edge kernels
// =====================================================================
__global__ void k_fill_f32(float* p, float v, int n) {
  int i = blockIdx.x * 256 + threadIdx.x;
  if (i < n) p[i] = v;
}
__global__ void k_fill_u32(unsigned* p, unsigned v, int n) {
  int i = blockIdx.x * 256 + threadIdx.x;
  if (i < n) p[i] = v;
}

// tax[n, 0:128] = norm_vals[n] * concat(taxo_mean[cats[col,0]], taxo_mean[cats[col,1]]), col = norm_cols[n]
__global__ void k_tax(const int* __restrict__ ncols, const float* __restrict__ nvals,
                      const int* __restrict__ cats, const float* __restrict__ tmean,
                      float* __restrict__ tax, int N) {
  int i = blockIdx.x * 256 + threadIdx.x;
  if (i >= N * 128) return;
  int n = i >> 7, c = i & 127;
  int col = ncols[n];
  int cat = cats[col * 2 + (c >> 6)];
  tax[i] = nvals[n] * tmean[cat * 64 + (c & 63)];
}

// wave per edge: wt[e] = dot(tax[src], tax[dst], 128); atomicMax over dst
__global__ void k_edge_wt(const int* __restrict__ src, const int* __restrict__ dst,
                          const float* __restrict__ tax, float* __restrict__ wt,
                          unsigned* __restrict__ imax, int E) {
  int lane = threadIdx.x & 31;
  int e = blockIdx.x * 8 + (threadIdx.x >> 5);
  if (e >= E) return;
  const float* ts = tax + (size_t)src[e] * 128;
  const float* td = tax + (size_t)dst[e] * 128;
  float acc = 0.0f;
  #pragma unroll
  for (int s = 0; s < 4; ++s) { int c = lane + 32 * s; acc += ts[c] * td[c]; }
  acc = wred(acc);
  if (lane == 0) { wt[e] = acc; atomicMax(imax + dst[e], f2ord(acc)); }
}

// sc[e] = exp(sc[e] - max[dst]); atomicAdd into sum[dst]
__global__ void k_edge_exp(float* __restrict__ sc, const int* __restrict__ dst,
                           const unsigned* __restrict__ imax, float* __restrict__ sum, int E) {
  int e = blockIdx.x * 256 + threadIdx.x;
  if (e >= E) return;
  int d = dst[e];
  float x = __expf(sc[e] - ord2f(imax[d]));
  sc[e] = x;
  atomicAdd(sum + d, x);
}

// sc[e] = coef * sc[e] / sum[dst[e]]
__global__ void k_alpha_scale(float* __restrict__ sc, const int* __restrict__ dst,
                              const float* __restrict__ sum, float coef, int E) {
  int e = blockIdx.x * 256 + threadIdx.x;
  if (e >= E) return;
  sc[e] = coef * sc[e] / sum[dst[e]];
}

// per-node attention coefficients: a_src = h.wh[:64], a_dst = h.wh[64:]
__global__ void k_node_ab(const float* __restrict__ h, const float* __restrict__ wh,
                          float* __restrict__ as_, float* __restrict__ ad_, int N) {
  int lane = threadIdx.x & 31;
  int n = blockIdx.x * 8 + (threadIdx.x >> 5);
  if (n >= N) return;
  float h0 = h[(size_t)n * 64 + lane], h1 = h[(size_t)n * 64 + lane + 32];
  float s0 = wred(h0 * wh[lane] + h1 * wh[lane + 32]);
  float s1 = wred(h0 * wh[64 + lane] + h1 * wh[96 + lane]);
  if (lane == 0) { as_[n] = s0; ad_[n] = s1; }
}

__global__ void k_edge_fscore(const int* __restrict__ src, const int* __restrict__ dst,
                              const float* __restrict__ as_, const float* __restrict__ ad_,
                              float* __restrict__ eF, unsigned* __restrict__ imax, int E) {
  int e = blockIdx.x * 256 + threadIdx.x;
  if (e >= E) return;
  float wf = as_[src[e]] + ad_[dst[e]];
  wf = (wf >= 0.0f) ? wf : SLOPE * wf;
  eF[e] = wf;
  atomicMax(imax + dst[e], f2ord(wf));
}

// wave per edge: alpha = ETA*ef/sumf[d] + alphat; z[d] += h[s]*alpha
__global__ void k_edge_scatter(const int* __restrict__ src, const int* __restrict__ dst,
                               const float* __restrict__ h, const float* __restrict__ ef,
                               const float* __restrict__ sumf, const float* __restrict__ alphat,
                               float* __restrict__ z, int E) {
  int lane = threadIdx.x & 31;
  int e = blockIdx.x * 8 + (threadIdx.x >> 5);
  if (e >= E) return;
  int s = src[e], d = dst[e];
  float alpha = ETA * ef[e] / sumf[d] + alphat[e];
  const float* hs = h + (size_t)s * 64;
  float* zd = z + (size_t)d * 64;
  atomicAdd(zd + lane,      hs[lane]      * alpha);
  atomicAdd(zd + lane + 32, hs[lane + 32] * alpha);
}

// h_rev[n,c] = idx_vals[n] * h[idx_cols[n], c]   (spmm with rows=arange)
__global__ void k_hrev(const int* __restrict__ icols, const float* __restrict__ ivals,
                       const float* __restrict__ h, float* __restrict__ out, int N) {
  int i = blockIdx.x * 256 + threadIdx.x;
  if (i >= N * 64) return;
  int n = i >> 6, c = i & 63;
  out[i] = ivals[n] * h[(size_t)icols[n] * 64 + c];
}

__global__ void k_gnorm(const int* __restrict__ ncols, const float* __restrict__ nvals,
                        const float* __restrict__ cur, float* __restrict__ out, int N) {
  int i = blockIdx.x * 256 + threadIdx.x;
  if (i >= N * 64) return;
  int n = i >> 6, c = i & 63;
  out[i] = nvals[n] * cur[(size_t)ncols[n] * 64 + c];
}

// wave per node: stpl = <tpsi[n], mi[n]> + psi_b; tau = sigmoid(stpl)*exp(-exp(logstd));
// cur = (1-tau)*hrev + tau*mi   (in place over hrev)
__global__ void k_tau(float* __restrict__ cur, const float* __restrict__ tpsi,
                      const int* __restrict__ cats, const float* __restrict__ tmean,
                      const float* __restrict__ tlogstd, const float* __restrict__ psib,
                      int which, int N) {
  int lane = threadIdx.x & 31;
  int n = blockIdx.x * 8 + (threadIdx.x >> 5);
  if (n >= N) return;
  int cat = cats[n * 2 + which];
  float mi0 = tmean[cat * 64 + lane], mi1 = tmean[cat * 64 + lane + 32];
  float t0 = tpsi[(size_t)n * 64 + lane], t1 = tpsi[(size_t)n * 64 + lane + 32];
  float stpl = wred(t0 * mi0 + t1 * mi1) + psib[0];
  float sig  = 1.0f / (1.0f + __expf(-stpl));
  float tau0 = sig * __expf(-__expf(tlogstd[cat * 64 + lane]));
  float tau1 = sig * __expf(-__expf(tlogstd[cat * 64 + lane + 32]));
  size_t b = (size_t)n * 64;
  cur[b + lane]      = (1.0f - tau0) * cur[b + lane]      + tau0 * mi0;
  cur[b + lane + 32] = (1.0f - tau1) * cur[b + lane + 32] + tau1 * mi1;
}

// wave per row: L2-normalize 32-col row of `pre`, write to out[n*96 + base + lane*stride]
__global__ void k_rownorm(const float* __restrict__ pre, float* __restrict__ out,
                          int basec, int stride, int N) {
  int lane = threadIdx.x & 31;
  int n = blockIdx.x * 8 + (threadIdx.x >> 5);
  if (n >= N) return;
  float v = pre[(size_t)n * 32 + lane];
  float ss = wred(v * v);
  float inv = 1.0f / fmaxf(sqrtf(ss), 1e-12f);
  out[(size_t)n * 96 + basec + lane * stride] = v * inv;
}

// =====================================================================
extern "C" void kernel_launch(void* const* d_in, const int* in_sizes, int n_in,
                              void* d_out, int out_size, void* d_ws, size_t ws_size,
                              hipStream_t stream) {
  (void)n_in; (void)out_size; (void)ws_size;
  const float* h_in    = (const float*)d_in[0];
  const int*   src     = (const int*)d_in[1];
  const int*   dst     = (const int*)d_in[2];
  const int*   cats    = (const int*)d_in[3];
  const int*   icols   = (const int*)d_in[5];
  const float* ivals   = (const float*)d_in[6];
  const int*   ncols   = (const int*)d_in[8];
  const float* nvals   = (const float*)d_in[9];
  const float* W_w     = (const float*)d_in[10];
  const float* W_b     = (const float*)d_in[11];
  const float* pa      = (const float*)d_in[12];
  const float* tmean   = (const float*)d_in[13];
  const float* tlogstd = (const float*)d_in[14];
  const float* wh      = (const float*)d_in[15];
  const float* cWw     = (const float*)d_in[16];
  const float* cWb     = (const float*)d_in[17];
  const float* psiW    = (const float*)d_in[18];
  const float* psib    = (const float*)d_in[19];
  const float* w0      = (const float*)d_in[20];
  const float* b0      = (const float*)d_in[21];
  const float* w1      = (const float*)d_in[22];
  const float* b1      = (const float*)d_in[23];
  const float* w2      = (const float*)d_in[24];
  const float* b2      = (const float*)d_in[25];

  const int N = in_sizes[0] / 64;
  const int E = in_sizes[1];

  char* base = (char*)d_ws; size_t off = 0;
  auto alloc = [&](size_t bytes) -> void* {
    void* p = base + off; off = (off + bytes + 255) & ~(size_t)255; return p;
  };
  float*    tax  = (float*)   alloc((size_t)N * 128 * sizeof(float));
  float*    eT   = (float*)   alloc((size_t)E * sizeof(float));   // wt -> et -> (1-eta)*alpha_t
  float*    eF   = (float*)   alloc((size_t)E * sizeof(float));   // wf -> ef
  float*    nA   = (float*)   alloc((size_t)N * 64 * sizeof(float));
  float*    nB   = (float*)   alloc((size_t)N * 64 * sizeof(float));
  float*    a_s  = (float*)   alloc((size_t)N * sizeof(float));
  float*    a_d  = (float*)   alloc((size_t)N * sizeof(float));
  unsigned* imax = (unsigned*)alloc((size_t)N * sizeof(unsigned));
  float*    fsum = (float*)   alloc((size_t)N * sizeof(float));
  float* hrev = tax;                    // tax region is free after alpha_t is built
  float* pre  = tax + (size_t)N * 64;
  float* raw  = (float*)d_out;          // first output (N x 64); also h after convs
  float* emb  = raw + (size_t)N * 64;   // second output (N x 96)

  dim3 b256(256);
  int gN   = (N + 255) / 256;
  int gE   = (E + 255) / 256;
  int gEw  = (E + 7) / 8;
  int gNw  = (N + 7) / 8;
  int gNC  = (N * 128 + 255) / 256;
  int gNH  = (N * 64 + 255) / 256;
  int gG   = (N + 127) / 128;

  // ---- taxonomy attention alpha_t (constant across both convs) ----
  k_fill_u32<<<gN, b256, 0, stream>>>(imax, ENC_NEG_INF, N);
  k_fill_f32<<<gN, b256, 0, stream>>>(fsum, 0.0f, N);
  k_tax<<<gNC, b256, 0, stream>>>(ncols, nvals, cats, tmean, tax, N);
  k_edge_wt<<<gEw, b256, 0, stream>>>(src, dst, tax, eT, imax, E);
  k_edge_exp<<<gE, b256, 0, stream>>>(eT, dst, imax, fsum, E);
  k_alpha_scale<<<gE, b256, 0, stream>>>(eT, dst, fsum, 1.0f - ETA, E);

  // ---- input projection + PReLU ----
  k_gemm64<64, true, true><<<gG, b256, 0, stream>>>(h_in, W_w, W_b, pa, nA, N);

  // ---- two conv layers ----
  for (int conv = 0; conv < 2; ++conv) {
    k_node_ab<<<gNw, b256, 0, stream>>>(nA, wh, a_s, a_d, N);
    k_fill_u32<<<gN, b256, 0, stream>>>(imax, ENC_NEG_INF, N);
    k_fill_f32<<<gN, b256, 0, stream>>>(fsum, 0.0f, N);
    k_fill_f32<<<gNH, b256, 0, stream>>>(nB, 0.0f, N * 64);
    k_edge_fscore<<<gE, b256, 0, stream>>>(src, dst, a_s, a_d, eF, imax, E);
    k_edge_exp<<<gE, b256, 0, stream>>>(eF, dst, imax, fsum, E);
    k_edge_scatter<<<gEw, b256, 0, stream>>>(src, dst, nA, eF, fsum, eT, nB, E);
    if (conv == 0)
      k_gemm64<64, true, true ><<<gG, b256, 0, stream>>>(nB, cWw, cWb, pa, nA, N);
    else
      k_gemm64<64, true, false><<<gG, b256, 0, stream>>>(nB, cWw, cWb, pa, raw, N);
  }

  // ---- emb_h = normalize(raw @ mlp_w2^T + b2) -> emb cols [0,32) ----
  k_gemm64<32, true, false><<<gG, b256, 0, stream>>>(raw, w2, b2, pa, pre, N);
  k_rownorm<<<gNw, b256, 0, stream>>>(pre, emb, 0, 1, N);

  // ---- two taxonomy-conditioned embeddings -> emb cols 32+2j+i ----
  for (int i = 0; i < 2; ++i) {
    k_hrev<<<gNH, b256, 0, stream>>>(icols, ivals, raw, hrev, N);
    k_gemm64<64, false, false><<<gG, b256, 0, stream>>>(hrev, psiW, nullptr, pa, nB, N);
    k_tau<<<gNw, b256, 0, stream>>>(hrev, nB, cats, tmean, tlogstd, psib, i, N);
    k_gnorm<<<gNH, b256, 0, stream>>>(ncols, nvals, hrev, nA, N);
    k_gemm64<32, true, false><<<gG, b256, 0, stream>>>(nA, i ? w1 : w0, i ? b1 : b0, pa, pre, N);
    k_rownorm<<<gNw, b256, 0, stream>>>(pre, emb, 32 + i, 2, N);
  }
}
